// multihead_self_attention_78288663871496
// MI455X (gfx1250) — matrix operand order repaired and checked by reading, not compiled, not run on hardware
//
#include <hip/hip_runtime.h>

// ---------------------------------------------------------------------------
// MI455X (gfx1250) multi-head self-attention with RoPE, bf16 WMMA pipeline.
//   B=4, S=2048, D_MODEL=1024, H=16, DK=64.  fp32 in/out, bf16 matmuls,
//   f32 accumulation, flash attention with async-to-LDS K/V staging.
// ---------------------------------------------------------------------------

#define DEV __device__ __forceinline__

constexpr int Bn = 4;
constexpr int Sn = 2048;
constexpr int Dn = 1024;
constexpr int Hn = 16;
constexpr int DKn = 64;

typedef __attribute__((ext_vector_type(8)))  float        v8f;
typedef __attribute__((ext_vector_type(16))) __bf16       v16bf;
typedef __attribute__((ext_vector_type(4)))  unsigned int u32x4;

union BF16x16 { v16bf v; u32x4 q[2]; };

// Load a 16-element bf16 WMMA operand from two 16-byte chunks.
static DEV v16bf load16(const __bf16* p0, const __bf16* p1) {
  BF16x16 u;
  u.q[0] = *reinterpret_cast<const u32x4*>(p0);
  u.q[1] = *reinterpret_cast<const u32x4*>(p1);
  return u.v;
}

static DEV v8f wmma_bf16(v16bf a, v16bf b, v8f c) {
  // (neg_a, A, neg_b, B, c_mod, C, reuse_a, reuse_b)
  return __builtin_amdgcn_wmma_f32_16x16x32_bf16(false, a, false, b,
                                                 (short)0, c, false, false);
}

// Async copy 16B global -> LDS (gfx1250, ASYNCcnt-tracked), GVS addressing.
static DEV void async_b128(unsigned lds_off, unsigned gbl_off, const __bf16* base) {
  asm volatile("global_load_async_to_lds_b128 %0, %1, %2"
               :: "v"(lds_off), "v"(gbl_off), "s"(base) : "memory");
}

// ---------------------------------------------------------------------------
// fp32 -> bf16 cast
// ---------------------------------------------------------------------------
__global__ __launch_bounds__(256) void cvt_bf16_kernel(const float* __restrict__ in,
                                                       __bf16* __restrict__ out, int n) {
  int i = blockIdx.x * blockDim.x + threadIdx.x;
  int stride = gridDim.x * blockDim.x;
  for (; i < n; i += stride) out[i] = (__bf16)in[i];
}

// ---------------------------------------------------------------------------
// C[m,n] = sum_k A[m,k] * W[n,k]   (A: MxK bf16, W: NxK bf16)
// One wave -> 32x64 output tile (8 WMMA / K-step); 8 waves -> 128x128 block.
// ---------------------------------------------------------------------------
__global__ __launch_bounds__(256) void gemm_bt_kernel(const __bf16* __restrict__ A,
                                                      const __bf16* __restrict__ W,
                                                      float* __restrict__ outF,
                                                      __bf16* __restrict__ outB,
                                                      int M, int N, int K) {
  const int wave = threadIdx.x >> 5;
  const int lane = threadIdx.x & 31;
  const int wm = wave >> 1, wn = wave & 1;
  const int m0 = blockIdx.y * 128 + wm * 32;
  const int n0 = blockIdx.x * 128 + wn * 64;
  const int lh = lane & 15;
  const int hi = lane >> 4;           // 0 for lanes 0-15, 1 for 16-31

  v8f c[2][4] = {{{}, {}, {}, {}}, {{}, {}, {}, {}}};

  // A operand lane base: row m, A-layout K offset (lanes>=16 carry K+8..)
  const __bf16* ap[2];
  ap[0] = A + (size_t)(m0 + lh) * K + hi * 8;
  ap[1] = A + (size_t)(m0 + 16 + lh) * K + hi * 8;
  // B operand lane base: column n == W row n; B-layout K offset (lanes>=16 carry K+8..)
  const __bf16* bp[4];
#pragma unroll
  for (int j = 0; j < 4; ++j)
    bp[j] = W + (size_t)(n0 + j * 16 + lh) * K + hi * 8;

  for (int k0 = 0; k0 < K; k0 += 32) {
    v16bf a0 = load16(ap[0] + k0, ap[0] + k0 + 16);
    v16bf a1 = load16(ap[1] + k0, ap[1] + k0 + 16);
    __builtin_prefetch(ap[0] + k0 + 128, 0, 1);  // global_prefetch_b8
#pragma unroll
    for (int j = 0; j < 4; ++j) {
      v16bf b = load16(bp[j] + k0, bp[j] + k0 + 16);
      c[0][j] = wmma_bf16(a0, b, c[0][j]);
      c[1][j] = wmma_bf16(a1, b, c[1][j]);
    }
  }

#pragma unroll
  for (int i = 0; i < 2; ++i)
#pragma unroll
    for (int j = 0; j < 4; ++j)
#pragma unroll
      for (int r = 0; r < 8; ++r) {
        const int m = m0 + i * 16 + r + hi * 8;
        const int n = n0 + j * 16 + lh;
        if (outF) outF[(size_t)m * N + n] = c[i][j][r];
        else      outB[(size_t)m * N + n] = (__bf16)c[i][j][r];
      }
}

// ---------------------------------------------------------------------------
// RoPE + head-major reorder: Qraw/Kraw [B,S,D] -> Qb/Kb [B*H, S, DK]
// ---------------------------------------------------------------------------
__global__ __launch_bounds__(256) void rope_kernel(const __bf16* __restrict__ Qraw,
                                                   const __bf16* __restrict__ Kraw,
                                                   __bf16* __restrict__ Qb,
                                                   __bf16* __restrict__ Kb) {
  const int total = Bn * Sn * Hn * (DKn / 2);   // 4,194,304
  int tid = blockIdx.x * blockDim.x + threadIdx.x;
  int stride = gridDim.x * blockDim.x;
  for (; tid < total; tid += stride) {
    const int i = tid & 31;
    const int h = (tid >> 5) & 15;
    const int s = (tid >> 9) & 2047;
    const int b = tid >> 20;
    const float inv = __expf(-(2.0f * (float)i / 64.0f) * 9.2103403719761836f);
    float sn, cs;
    __sincosf((float)s * inv, &sn, &cs);

    const size_t src = (size_t)(b * Sn + s) * Dn + h * DKn + 2 * i;
    const size_t dst = ((size_t)(b * Hn + h) * Sn + s) * DKn + 2 * i;

    float q1 = (float)Qraw[src], q2 = (float)Qraw[src + 1];
    Qb[dst]     = (__bf16)(q1 * cs - q2 * sn);
    Qb[dst + 1] = (__bf16)(q1 * sn + q2 * cs);

    float k1 = (float)Kraw[src], k2 = (float)Kraw[src + 1];
    Kb[dst]     = (__bf16)(k1 * cs - k2 * sn);
    Kb[dst + 1] = (__bf16)(k1 * sn + k2 * cs);
  }
}

// V transpose: Vraw [B,S,D] -> Vt [B*H, DK, S]
__global__ __launch_bounds__(256) void vtrans_kernel(const __bf16* __restrict__ Vraw,
                                                     __bf16* __restrict__ Vt) {
  const int total = Bn * Hn * DKn * Sn;          // 8,388,608
  int tid = blockIdx.x * blockDim.x + threadIdx.x;
  int stride = gridDim.x * blockDim.x;
  for (; tid < total; tid += stride) {
    const int s  = tid & 2047;
    const int dk = (tid >> 11) & 63;
    const int h  = (tid >> 17) & 15;
    const int b  = tid >> 21;
    Vt[tid] = Vraw[(size_t)(b * Sn + s) * Dn + h * DKn + dk];
  }
}

// ---------------------------------------------------------------------------
// Causal flash attention. 4 waves/block, 64 queries/block (16 per wave).
//   Qb/Kb: [B*H, S, 64] bf16,  Vt: [B*H, 64, S] bf16,  Ob: [B, S, 1024] bf16.
// K/V tiles staged into LDS with global_load_async_to_lds_b128 (double
// buffered, shared by all 4 waves); scores + P*V via WMMA; the C->A layout
// transpose of P goes through a per-wave LDS tile.
// ---------------------------------------------------------------------------
__global__ __launch_bounds__(128) void attn_kernel(const __bf16* __restrict__ Qb,
                                                   const __bf16* __restrict__ Kb,
                                                   const __bf16* __restrict__ Vt,
                                                   __bf16* __restrict__ Ob) {
  __shared__ __align__(16) __bf16 ktile[2][32][64];   // [buf][key][dk]   8KB
  __shared__ __align__(16) __bf16 vtile[2][64][32];   // [buf][dk][key]   8KB
  __shared__ __align__(16) __bf16 plds[4][16 * 32];   // P tile per wave  4KB

  const int tid  = threadIdx.x;
  const int wave = tid >> 5;
  const int lane = tid & 31;
  const int lh = lane & 15;
  const int hi = lane >> 4;
  const int bh = blockIdx.y;
  const int b = bh >> 4, h = bh & 15;
  const int q0_blk = blockIdx.x * 64;
  const int q0 = q0_blk + wave * 16;
  const int kmax = q0_blk + 63;            // block-uniform causal bound

  const __bf16* Qh = Qb + (size_t)bh * Sn * DKn;
  const __bf16* Kh = Kb + (size_t)bh * Sn * DKn;
  const __bf16* Vh = Vt + (size_t)bh * DKn * Sn;
  __bf16* pw = &plds[wave][0];

  // Q A-operands held in registers for the whole key loop
  const __bf16* qrow = Qh + (size_t)(q0 + lh) * DKn + hi * 16;
  const v16bf qa0 = load16(qrow,      qrow + 8);    // dk  0..31
  const v16bf qa1 = load16(qrow + 32, qrow + 40);   // dk 32..63

  float m_r[8], l_r[8];
  v8f oacc[4] = {{}, {}, {}, {}};
#pragma unroll
  for (int r = 0; r < 8; ++r) { m_r[r] = -1e30f; l_r[r] = 0.0f; }

  const float scale = 0.125f;   // 1/sqrt(64)

  // --- cooperative async stage of one 32-key K/V tile (4 ops/thread) ---
  auto stage = [&](int buf, int k0) {
#pragma unroll
    for (int it = 0; it < 2; ++it) {          // K: 32 rows x 128B
      const int ch = tid + it * 128;          // 256 chunks of 16B
      const int row = ch >> 3, off = (ch & 7) * 8;
      async_b128((unsigned)(size_t)&ktile[buf][row][off],
                 (unsigned)(((k0 + row) * DKn + off) * 2), Kh);
    }
#pragma unroll
    for (int it = 0; it < 2; ++it) {          // V: 64 rows x 64B
      const int ch = tid + it * 128;
      const int row = ch >> 2, off = (ch & 3) * 8;
      async_b128((unsigned)(size_t)&vtile[buf][row][off],
                 (unsigned)((row * Sn + k0 + off) * 2), Vh);
    }
  };

  stage(0, 0);

  for (int k0 = 0; k0 <= kmax; k0 += 32) {
    const int buf = (k0 >> 5) & 1;
    __syncthreads();                               // tile buf^1 readers done
    const bool have_next = (k0 + 32 <= kmax);
    if (have_next) {
      stage(buf ^ 1, k0 + 32);
      asm volatile("s_wait_asynccnt 0x4" ::: "memory");  // tile buf complete
    } else {
      asm volatile("s_wait_asynccnt 0x0" ::: "memory");
    }
    __syncthreads();                               // tile buf visible to all

    if (k0 <= q0 + 15) {                           // wave-uniform causal skip
      // ---- scores: two 16x16 tiles from the LDS K tile ----
      const __bf16* kr0 = &ktile[buf][lh][hi * 16];
      const __bf16* kr1 = &ktile[buf][16 + lh][hi * 16];
      v16bf kb0lo = load16(kr0,      kr0 + 8);     // keys 0..15,  dk  0..31
      v16bf kb0hi = load16(kr0 + 32, kr0 + 40);    //              dk 32..63
      v16bf kb1lo = load16(kr1,      kr1 + 8);     // keys 16..31
      v16bf kb1hi = load16(kr1 + 32, kr1 + 40);

      v8f c0 = {}, c1 = {};
      c0 = wmma_bf16(qa0, kb0lo, c0);
      c0 = wmma_bf16(qa1, kb0hi, c0);
      c1 = wmma_bf16(qa0, kb1lo, c1);
      c1 = wmma_bf16(qa1, kb1hi, c1);

      asm volatile("s_wait_dscnt 0x0" ::: "memory");   // P readers of prev iter

      // ---- online softmax update over both 16-key subtiles ----
      float corr[8];
#pragma unroll
      for (int r = 0; r < 8; ++r) {
        const int m = r + hi * 8;
        const bool msk0 = (k0 + lh)      > (q0 + m);
        const bool msk1 = (k0 + 16 + lh) > (q0 + m);
        float s0 = msk0 ? -1e30f : c0[r] * scale;
        float s1 = msk1 ? -1e30f : c1[r] * scale;

        float mx = fmaxf(s0, s1);
        mx = fmaxf(mx, __shfl_xor(mx, 1, 16));
        mx = fmaxf(mx, __shfl_xor(mx, 2, 16));
        mx = fmaxf(mx, __shfl_xor(mx, 4, 16));
        mx = fmaxf(mx, __shfl_xor(mx, 8, 16));

        const float m_new = fmaxf(m_r[r], mx);
        const float cr = __expf(m_r[r] - m_new);
        const float p0 = msk0 ? 0.0f : __expf(s0 - m_new);
        const float p1 = msk1 ? 0.0f : __expf(s1 - m_new);

        float rs = p0 + p1;
        rs += __shfl_xor(rs, 1, 16);
        rs += __shfl_xor(rs, 2, 16);
        rs += __shfl_xor(rs, 4, 16);
        rs += __shfl_xor(rs, 8, 16);

        l_r[r] = l_r[r] * cr + rs;
        m_r[r] = m_new;
        corr[r] = cr;

        pw[m * 32 + lh]      = (__bf16)p0;     // ds_store_b16 (C->A transpose)
        pw[m * 32 + 16 + lh] = (__bf16)p1;
      }

#pragma unroll
      for (int n = 0; n < 4; ++n)
#pragma unroll
        for (int r = 0; r < 8; ++r) oacc[n][r] *= corr[r];

      asm volatile("s_wait_dscnt 0x0" ::: "memory");   // P stores visible
      const __bf16* prow = pw + lh * 32 + hi * 16;
      v16bf pa = load16(prow, prow + 8);               // P in A-matrix layout

      // ---- oacc += P x V (V from LDS tile) ----
#pragma unroll
      for (int n = 0; n < 4; ++n) {
        const __bf16* vr = &vtile[buf][n * 16 + lh][hi * 16];
        v16bf vb = load16(vr, vr + 8);
        oacc[n] = wmma_bf16(pa, vb, oacc[n]);
      }
    }
  }

  // ---- normalize and write O in [B, S, D] layout ----
#pragma unroll
  for (int n = 0; n < 4; ++n)
#pragma unroll
    for (int r = 0; r < 8; ++r) {
      const int m = r + hi * 8;
      const float o = oacc[n][r] / l_r[r];
      Ob[((size_t)b * Sn + q0 + m) * Dn + h * DKn + n * 16 + lh] = (__bf16)o;
    }
}

// ---------------------------------------------------------------------------
// Host-side orchestration
// ---------------------------------------------------------------------------
extern "C" void kernel_launch(void* const* d_in, const int* in_sizes, int n_in,
                              void* d_out, int out_size, void* d_ws, size_t ws_size,
                              hipStream_t stream) {
  (void)in_sizes; (void)n_in; (void)out_size; (void)ws_size;

  const float* x  = (const float*)d_in[0];
  const float* Wq = (const float*)d_in[1];
  const float* Wk = (const float*)d_in[2];
  const float* Wv = (const float*)d_in[3];
  const float* Wo = (const float*)d_in[4];
  float* out = (float*)d_out;

  char* w = (char*)d_ws;
  constexpr size_t MB = 1ull << 20;
  __bf16* xb   = (__bf16*)(w + 0);        // 16 MB : x bf16        [8192,1024]
  __bf16* wqb  = (__bf16*)(w + 16 * MB);  //  2 MB
  __bf16* wkb  = (__bf16*)(w + 18 * MB);  //  2 MB
  __bf16* wvb  = (__bf16*)(w + 20 * MB);  //  2 MB
  __bf16* wob  = (__bf16*)(w + 22 * MB);  //  2 MB
  __bf16* qraw = (__bf16*)(w + 24 * MB);  // 16 MB : Q pre-rope
  __bf16* kraw = (__bf16*)(w + 40 * MB);  // 16 MB
  __bf16* vraw = (__bf16*)(w + 56 * MB);  // 16 MB
  __bf16* qb   = (__bf16*)(w + 72 * MB);  // 16 MB : [B*H, S, 64]
  __bf16* kb   = (__bf16*)(w + 88 * MB);  // 16 MB : [B*H, S, 64]
  __bf16* vt   = (__bf16*)(w + 104 * MB); // 16 MB : [B*H, 64, S]
  __bf16* ob   = (__bf16*)(w + 24 * MB);  // alias qraw (dead after rope)

  const int nX = Bn * Sn * Dn;   // 8,388,608
  const int nW = Dn * Dn;        // 1,048,576

  // 1) casts to bf16
  cvt_bf16_kernel<<<nX / 256, 256, 0, stream>>>(x,  xb,  nX);
  cvt_bf16_kernel<<<nW / 256, 256, 0, stream>>>(Wq, wqb, nW);
  cvt_bf16_kernel<<<nW / 256, 256, 0, stream>>>(Wk, wkb, nW);
  cvt_bf16_kernel<<<nW / 256, 256, 0, stream>>>(Wv, wvb, nW);
  cvt_bf16_kernel<<<nW / 256, 256, 0, stream>>>(Wo, wob, nW);

  // 2) QKV projections (WMMA, 128x128 block tiles)
  dim3 ggrid(Dn / 128, (Bn * Sn) / 128);   // (8, 64)
  gemm_bt_kernel<<<ggrid, 256, 0, stream>>>(xb, wqb, nullptr, qraw, Bn * Sn, Dn, Dn);
  gemm_bt_kernel<<<ggrid, 256, 0, stream>>>(xb, wkb, nullptr, kraw, Bn * Sn, Dn, Dn);
  gemm_bt_kernel<<<ggrid, 256, 0, stream>>>(xb, wvb, nullptr, vraw, Bn * Sn, Dn, Dn);

  // 3) RoPE + head-major reorder; V transpose
  rope_kernel<<<(Bn * Sn * Hn * 32) / 256, 256, 0, stream>>>(qraw, kraw, qb, kb);
  vtrans_kernel<<<(Bn * Hn * DKn * Sn) / 256, 256, 0, stream>>>(vraw, vt);

  // 4) causal flash attention (WMMA + async-to-LDS staging)
  attn_kernel<<<dim3(Sn / 64, Bn * Hn), 128, 0, stream>>>(qb, kb, vt, ob);

  // 5) output projection -> fp32 d_out (WMMA)
  gemm_bt_kernel<<<ggrid, 256, 0, stream>>>(ob, wob, out, nullptr, Bn * Sn, Dn, Dn);
}
